// STSLateFusionGTFlow_83056077570755
// MI455X (gfx1250) — matrix operand order repaired and checked by reading, not compiled, not run on hardware
//
#include <hip/hip_runtime.h>
#include <hip/hip_bf16.h>
#include <cstddef>

// ---------------------------------------------------------------------------
// Problem dims (B=1): images 3x288x624; features 32x96x208; cost D=64,
// 3D volume (ch, d=64, h=96, w=208); refinement U-Net on 288x624.
// ---------------------------------------------------------------------------

typedef __attribute__((ext_vector_type(16))) _Float16 v16h;
typedef __attribute__((ext_vector_type(8)))  _Float16 v8h;
typedef __attribute__((ext_vector_type(8)))  float    v8f;

__device__ __forceinline__ v16h zero_v16h() {
  v16h z;
#pragma unroll
  for (int i = 0; i < 16; ++i) z[i] = (_Float16)0.0f;
  return z;
}

__device__ __forceinline__ int clampi(int v, int lo, int hi) {
  return v < lo ? lo : (v > hi ? hi : v);
}

// ---------------------------------------------------------------------------
// Feature extraction conv1: 5x5 stride 3, 3->32, SAME (pad_beg=1), ReLU.
// in: [3][288][624] f32, out: [32][96][208] f32
// ---------------------------------------------------------------------------
__global__ void feat_conv1(const float* __restrict__ img,
                           const float* __restrict__ w,
                           const float* __restrict__ b,
                           float* __restrict__ out) {
  int idx = blockIdx.x * blockDim.x + threadIdx.x;
  if (idx >= 32 * 96 * 208) return;
  int x = idx % 208;
  int t = idx / 208;
  int y = t % 96;
  int co = t / 96;
  float acc = b[co];
  for (int kh = 0; kh < 5; ++kh) {
    int iy = y * 3 + kh - 1;
    if ((unsigned)iy >= 288u) continue;
    for (int kw = 0; kw < 5; ++kw) {
      int ix = x * 3 + kw - 1;
      if ((unsigned)ix >= 624u) continue;
      for (int ci = 0; ci < 3; ++ci) {
        acc += img[(ci * 288 + iy) * 624 + ix] * w[((co * 3 + ci) * 5 + kh) * 5 + kw];
      }
    }
  }
  out[idx] = fmaxf(acc, 0.0f);
}

// ---------------------------------------------------------------------------
// Generic 3x3 conv2d, NCHW f32, optional stride-2 (pad 0), optional fused
// nearest-up2 on the input (stride 1, pad 1), optional ReLU.
// ---------------------------------------------------------------------------
__global__ void conv2d_generic(const float* __restrict__ in,
                               const float* __restrict__ w,
                               const float* __restrict__ b,
                               float* __restrict__ out,
                               int Cin, int Cout, int Hin, int Win,
                               int Hout, int Wout, int stride, int pad,
                               int up, int relu) {
  int idx = blockIdx.x * blockDim.x + threadIdx.x;
  int total = Cout * Hout * Wout;
  if (idx >= total) return;
  int wo = idx % Wout;
  int t  = idx / Wout;
  int ho = t % Hout;
  int co = t / Hout;
  float acc = b[co];
  const int Hl = up ? (Hin * 2) : Hin;
  const int Wl = up ? (Win * 2) : Win;
  for (int kh = 0; kh < 3; ++kh) {
    int ih = ho * stride + kh - pad;
    if ((unsigned)ih >= (unsigned)Hl) continue;
    int ihs = up ? (ih >> 1) : ih;
    for (int kw = 0; kw < 3; ++kw) {
      int iw = wo * stride + kw - pad;
      if ((unsigned)iw >= (unsigned)Wl) continue;
      int iws = up ? (iw >> 1) : iw;
      const float* ip = in + (size_t)ihs * Win + iws;
      const float* wp = w + ((size_t)(co * Cin) * 3 + kh) * 3 + kw;
      for (int ci = 0; ci < Cin; ++ci) {
        acc += ip[(size_t)ci * Hin * Win] * wp[(size_t)ci * 9];
      }
    }
  }
  if (relu) acc = fmaxf(acc, 0.0f);
  out[idx] = acc;
}

// ---------------------------------------------------------------------------
// Feature conv3: 3x3 32->32 ReLU, converts to f16 in [h][w][c] layout
// (channel-innermost so WMMA B fragments are single contiguous 32B loads).
// ---------------------------------------------------------------------------
__global__ void feat_conv3_f16(const float* __restrict__ in,
                               const float* __restrict__ w,
                               const float* __restrict__ b,
                               _Float16* __restrict__ outf) {
  int idx = blockIdx.x * blockDim.x + threadIdx.x;
  if (idx >= 32 * 96 * 208) return;
  int x = idx % 208;
  int t = idx / 208;
  int y = t % 96;
  int co = t / 96;
  float acc = b[co];
  for (int kh = 0; kh < 3; ++kh) {
    int iy = y + kh - 1;
    if ((unsigned)iy >= 96u) continue;
    for (int kw = 0; kw < 3; ++kw) {
      int ix = x + kw - 1;
      if ((unsigned)ix >= 208u) continue;
      for (int ci = 0; ci < 32; ++ci) {
        acc += in[(ci * 96 + iy) * 208 + ix] * w[((co * 32 + ci) * 3 + kh) * 3 + kw];
      }
    }
  }
  outf[((size_t)y * 208 + x) * 32 + co] = (_Float16)fmaxf(acc, 0.0f);
}

// ---------------------------------------------------------------------------
// Weight repack into a WMMA A-matrix (16x32 f16) per-lane layout (contiguous K):
//   lane<16: M=lane, half hh -> K = hh
//   lane>=16: M=lane-16,      K = hh + 16
// conv1: wA1[tap(27)][cblk(2)][lane(32)][half(16)], K-local = in-channel.
// ---------------------------------------------------------------------------
__global__ void repack_w1(const float* __restrict__ w, _Float16* __restrict__ wA) {
  int idx = blockIdx.x * blockDim.x + threadIdx.x;
  if (idx >= 27 * 2 * 32 * 16) return;
  const int hh   = idx & 15;
  const int lane = (idx >> 4) & 31;
  const int cblk = (idx >> 9) & 1;
  const int tap  = idx >> 10;
  const int o  = lane & 15;
  const int kl = hh + ((lane >= 16) ? 16 : 0);
  const int ci = cblk * 32 + kl;
  // m_w1: OIDHW (16,64,3,3,3); flat tap = dd*9+dh*3+dw
  wA[idx] = (_Float16)w[((size_t)o * 64 + ci) * 27 + tap];
}

// conv2: K = tap-pair x 16 in-ch. wA2[kb(14)][lane(32)][half(16)].
// K-local <16 -> tap 2*kb, in-ch kl;  >=16 -> tap 2*kb+1, in-ch kl-16.
// Pad tap 27 gets zero weights.
__global__ void repack_w2(const float* __restrict__ w, _Float16* __restrict__ wA) {
  int idx = blockIdx.x * blockDim.x + threadIdx.x;
  if (idx >= 14 * 32 * 16) return;
  const int hh   = idx & 15;
  const int lane = (idx >> 4) & 31;
  const int kb   = idx >> 9;
  const int o  = lane & 15;
  const int kl = hh + ((lane >= 16) ? 16 : 0);
  const int tap = 2 * kb + (kl >= 16 ? 1 : 0);
  const int ci  = kl & 15;
  wA[idx] = (tap < 27) ? (_Float16)w[((size_t)o * 16 + ci) * 27 + tap] : (_Float16)0.0f;
}

// ---------------------------------------------------------------------------
// 3D conv1 over the (virtual) cost volume, fused cost construction.
//   cost[c<32, d,h,w]  = l0f[h,w,c]        if w-d>=0 else 0
//   cost[c>=32, d,h,w] = r0f[h,w-d,c-32]   if w-d>=0 else 0
// GEMM: M=16 out-ch, K = 64ch x 27 taps. Each wave computes a 16x64 macro-
// tile: 4 accumulators over 4 consecutive h rows sharing one W-tile, so each
// pair of A (weight) fragments feeds 8 WMMAs. Validity handled branchlessly:
// unconditional load from clamped in-bounds address + vector cndmask to zero.
// B layout (contiguous K, as in wA): lanes 0-15 hold K=0..15 (N=lane), lanes 16-31 hold K=16..31.
// Output h1: f16 [d][h][w][16].
// ---------------------------------------------------------------------------
__global__ __launch_bounds__(256)
void cost_conv1_wmma(const _Float16* __restrict__ l0f,
                     const _Float16* __restrict__ r0f,
                     const _Float16* __restrict__ wA,
                     const float* __restrict__ bias,
                     _Float16* __restrict__ h1) {
  const int lane = threadIdx.x & 31;
  const int tile = blockIdx.x * 8 + (threadIdx.x >> 5);  // 64*24*13 = 19968 groups
  const int n  = lane & 15;
  const int hi = lane >> 4;
  const int wt = tile % 13;
  const int t2 = tile / 13;
  const int h0 = (t2 % 24) * 4;
  const int d  = t2 / 24;
  const int wn = wt * 16 + n;

  v8f accs[4];
#pragma unroll
  for (int j = 0; j < 4; ++j)
#pragma unroll
    for (int v = 0; v < 8; ++v) accs[j][v] = 0.0f;
  const v16h bz = zero_v16h();

  for (int dd = 0; dd < 3; ++dd) {
    const int  dz  = d + dd - 1;
    const bool zok = (unsigned)dz < 64u;
    for (int dh = 0; dh < 3; ++dh) {
#pragma unroll
      for (int dw = 0; dw < 3; ++dw) {
        const int  tap = (dd * 3 + dh) * 3 + dw;
        const int  wx  = wn + dw - 1;
        const int  wr  = wx - dz;
        const bool vd  = zok && ((unsigned)wx < 208u) && (wr >= 0);
        const int  wxc = clampi(wx, 0, 207);
        const int  wrc = clampi(wr, 0, 207);
        // Weight fragments for this tap (channels 0..31 | 32..63), reused x4
        const v16h a0 = *(const v16h*)(wA + (size_t)((tap * 2 + 0) * 32 + lane) * 16);
        const v16h a1 = *(const v16h*)(wA + (size_t)((tap * 2 + 1) * 32 + lane) * 16);
#pragma unroll
        for (int j = 0; j < 4; ++j) {
          const int  hy  = h0 + j + dh - 1;
          const bool ok  = vd && ((unsigned)hy < 96u);
          const int  hyc = clampi(hy, 0, 95);
          v16h bl = *(const v16h*)(l0f + ((size_t)hyc * 208 + wxc) * 32 + hi * 16);
          v16h br = *(const v16h*)(r0f + ((size_t)hyc * 208 + wrc) * 32 + hi * 16);
          bl = ok ? bl : bz;
          br = ok ? br : bz;
          accs[j] = __builtin_amdgcn_wmma_f32_16x16x32_f16(false, a0, false, bl,
                                                           (short)0, accs[j], false, false);
          accs[j] = __builtin_amdgcn_wmma_f32_16x16x32_f16(false, a1, false, br,
                                                           (short)0, accs[j], false, false);
        }
      }
    }
  }

  // D layout: lane<16: N=lane, VGPR v -> M=v; lane>=16: M=v+8. Bias+ReLU, f16.
  const int m0 = hi * 8;
  float bb[8];
#pragma unroll
  for (int v = 0; v < 8; ++v) bb[v] = bias[m0 + v];
#pragma unroll
  for (int j = 0; j < 4; ++j) {
    v8h o;
#pragma unroll
    for (int v = 0; v < 8; ++v) {
      o[v] = (_Float16)fmaxf(accs[j][v] + bb[v], 0.0f);
    }
    *(v8h*)(h1 + (((size_t)d * 96 + (h0 + j)) * 208 + wn) * 16 + m0) = o;
  }
}

// ---------------------------------------------------------------------------
// 3D conv2: 16->16 ch, K = 27 taps x 16 ch padded to 14 K-blocks of 32
// (2 taps per WMMA step; lanes 0-15 read tap t0's voxel, lanes 16-31 tap t1's).
// Same 16x64 macro-tile: one A fragment feeds 4 WMMAs.
// ---------------------------------------------------------------------------
__global__ __launch_bounds__(256)
void cost_conv2_wmma(const _Float16* __restrict__ h1,
                     const _Float16* __restrict__ wA,
                     const float* __restrict__ bias,
                     _Float16* __restrict__ h2) {
  const int lane = threadIdx.x & 31;
  const int tile = blockIdx.x * 8 + (threadIdx.x >> 5);  // 19968 groups
  const int n  = lane & 15;
  const int hi = lane >> 4;
  const int wt = tile % 13;
  const int t2 = tile / 13;
  const int h0 = (t2 % 24) * 4;
  const int d  = t2 / 24;
  const int wn = wt * 16 + n;

  v8f accs[4];
#pragma unroll
  for (int j = 0; j < 4; ++j)
#pragma unroll
    for (int v = 0; v < 8; ++v) accs[j][v] = 0.0f;
  const v16h bz = zero_v16h();

  for (int kb = 0; kb < 14; ++kb) {
    int tsel = 2 * kb + hi;
    if (tsel > 26) tsel = 26;  // pad K-half; its weights are zero
    const int dd  = tsel / 9;
    const int rem = tsel - dd * 9;
    const int dh  = rem / 3;
    const int dw  = rem - dh * 3;
    const int  dz  = d + dd - 1;
    const int  wx  = wn + dw - 1;
    const bool vd  = ((unsigned)dz < 64u) && ((unsigned)wx < 208u);
    const int  dzc = clampi(dz, 0, 63);
    const int  wxc = clampi(wx, 0, 207);
    const v16h a = *(const v16h*)(wA + (size_t)(kb * 32 + lane) * 16);
#pragma unroll
    for (int j = 0; j < 4; ++j) {
      const int  hy  = h0 + j + dh - 1;
      const bool ok  = vd && ((unsigned)hy < 96u);
      const int  hyc = clampi(hy, 0, 95);
      v16h b = *(const v16h*)(h1 + (((size_t)dzc * 96 + hyc) * 208 + wxc) * 16);
      b = ok ? b : bz;
      accs[j] = __builtin_amdgcn_wmma_f32_16x16x32_f16(false, a, false, b,
                                                       (short)0, accs[j], false, false);
    }
  }

  const int m0 = hi * 8;
  float bb[8];
#pragma unroll
  for (int v = 0; v < 8; ++v) bb[v] = bias[m0 + v];
#pragma unroll
  for (int j = 0; j < 4; ++j) {
    v8h o;
#pragma unroll
    for (int v = 0; v < 8; ++v) {
      o[v] = (_Float16)fmaxf(accs[j][v] + bb[v], 0.0f);
    }
    *(v8h*)(h2 + (((size_t)d * 96 + (h0 + j)) * 208 + wn) * 16 + m0) = o;
  }
}

// ---------------------------------------------------------------------------
// 3D conv3: 16 -> 1 channel (no activation). Direct; ~1.1 GFLOP.
// out cost_out: f32 [d][h][w]
// ---------------------------------------------------------------------------
__global__ void cost_conv3(const _Float16* __restrict__ h2,
                           const float* __restrict__ w,
                           const float* __restrict__ b,
                           float* __restrict__ co) {
  int idx = blockIdx.x * blockDim.x + threadIdx.x;
  if (idx >= 64 * 96 * 208) return;
  int x = idx % 208;
  int t = idx / 208;
  int y = t % 96;
  int dz = t / 96;
  float acc = b[0];
  for (int dd = 0; dd < 3; ++dd) {
    int z = dz + dd - 1;
    if ((unsigned)z >= 64u) continue;
    for (int dh = 0; dh < 3; ++dh) {
      int yy = y + dh - 1;
      if ((unsigned)yy >= 96u) continue;
      for (int dw = 0; dw < 3; ++dw) {
        int xx = x + dw - 1;
        if ((unsigned)xx >= 208u) continue;
        const _Float16* vox = h2 + (((size_t)z * 96 + yy) * 208 + xx) * 16;
        const int tap = (dd * 3 + dh) * 3 + dw;
        for (int c = 0; c < 16; ++c) {
          acc += (float)vox[c] * w[c * 27 + tap];
        }
      }
    }
  }
  co[idx] = acc;
}

// Softmax over d (64) + disparity expectation -> disp_low [96][208]
__global__ void softmax_expect(const float* __restrict__ co,
                               float* __restrict__ dl) {
  int idx = blockIdx.x * blockDim.x + threadIdx.x;
  if (idx >= 96 * 208) return;
  float mx = -1e30f;
  for (int d = 0; d < 64; ++d) mx = fmaxf(mx, co[(size_t)d * 19968 + idx]);
  float s = 0.0f, ws = 0.0f;
  for (int d = 0; d < 64; ++d) {
    float e = expf(co[(size_t)d * 19968 + idx] - mx);
    s += e;
    ws += e * (float)d;
  }
  dl[idx] = ws / s;
}

// 3x bilinear upsample (half-pixel centers, edge clamp) and scale by 3.0
__global__ void upsample3_linear(const float* __restrict__ dl,
                                 float* __restrict__ d0) {
  int idx = blockIdx.x * blockDim.x + threadIdx.x;
  if (idx >= 288 * 624) return;
  int x = idx % 624;
  int y = idx / 624;
  float sx = (x + 0.5f) * (1.0f / 3.0f) - 0.5f;
  float sy = (y + 0.5f) * (1.0f / 3.0f) - 0.5f;
  float fx = floorf(sx), fy = floorf(sy);
  float tx = sx - fx, ty = sy - fy;
  int x0 = (int)fx, y0 = (int)fy;
  int x0c = clampi(x0, 0, 207);
  int x1c = clampi(x0 + 1, 0, 207);
  int y0c = clampi(y0, 0, 95);
  int y1c = clampi(y0 + 1, 0, 95);
  float v00 = dl[y0c * 208 + x0c], v01 = dl[y0c * 208 + x1c];
  float v10 = dl[y1c * 208 + x0c], v11 = dl[y1c * 208 + x1c];
  float v = (1.0f - ty) * ((1.0f - tx) * v00 + tx * v01) +
            ty * ((1.0f - tx) * v10 + tx * v11);
  d0[idx] = 3.0f * v;
}

// pack [disp0, d1_gt] -> x2 (2,288,624)
__global__ void pack2(const float* __restrict__ d0,
                      const float* __restrict__ d1,
                      float* __restrict__ x2) {
  int idx = blockIdx.x * blockDim.x + threadIdx.x;
  if (idx >= 288 * 624) return;
  x2[idx] = d0[idx];
  x2[288 * 624 + idx] = d1[idx];
}

__global__ void add_out(const float* __restrict__ refined,
                        const float* __restrict__ d0,
                        float* __restrict__ out) {
  int idx = blockIdx.x * blockDim.x + threadIdx.x;
  if (idx >= 288 * 624) return;
  out[idx] = refined[idx] + d0[idx];
}

// ---------------------------------------------------------------------------
// Host-side orchestration
// ---------------------------------------------------------------------------
extern "C" void kernel_launch(void* const* d_in, const int* in_sizes, int n_in,
                              void* d_out, int out_size, void* d_ws, size_t ws_size,
                              hipStream_t stream) {
  const float* l0_img = (const float*)d_in[0];
  const float* r0_img = (const float*)d_in[1];
  const float* d1_gt  = (const float*)d_in[2];
  const float* f_w1 = (const float*)d_in[3];  const float* f_b1 = (const float*)d_in[4];
  const float* f_w2 = (const float*)d_in[5];  const float* f_b2 = (const float*)d_in[6];
  const float* f_w3 = (const float*)d_in[7];  const float* f_b3 = (const float*)d_in[8];
  const float* m_w1 = (const float*)d_in[9];  const float* m_b1 = (const float*)d_in[10];
  const float* m_w2 = (const float*)d_in[11]; const float* m_b2 = (const float*)d_in[12];
  const float* m_w3 = (const float*)d_in[13]; const float* m_b3 = (const float*)d_in[14];
  const float* r_ew1 = (const float*)d_in[15]; const float* r_eb1 = (const float*)d_in[16];
  const float* r_ew2 = (const float*)d_in[17]; const float* r_eb2 = (const float*)d_in[18];
  const float* r_ew3 = (const float*)d_in[19]; const float* r_eb3 = (const float*)d_in[20];
  const float* r_dw1 = (const float*)d_in[21]; const float* r_db1 = (const float*)d_in[22];
  const float* r_dw2 = (const float*)d_in[23]; const float* r_db2 = (const float*)d_in[24];
  const float* r_dw3 = (const float*)d_in[25]; const float* r_db3 = (const float*)d_in[26];
  const float* r_ow  = (const float*)d_in[27]; const float* r_ob  = (const float*)d_in[28];
  float* out = (float*)d_out;
  (void)in_sizes; (void)n_in; (void)out_size; (void)ws_size;

  // Workspace carve-up (~115 MB total)
  char* base = (char*)d_ws;
  size_t off = 0;
  auto alloc = [&](size_t bytes) -> char* {
    char* p = base + off;
    off = (off + bytes + 255) & ~(size_t)255;
    return p;
  };
  float*    fA       = (float*)alloc((size_t)638976 * 4);          // 32x96x208 f32
  float*    fB       = (float*)alloc((size_t)638976 * 4);
  _Float16* l0f      = (_Float16*)alloc((size_t)638976 * 2);       // [h][w][32] f16
  _Float16* r0f      = (_Float16*)alloc((size_t)638976 * 2);
  _Float16* wA1      = (_Float16*)alloc((size_t)27648 * 2);        // 27x2x32x16
  _Float16* wA2      = (_Float16*)alloc((size_t)7168 * 2);         // 14x32x16
  _Float16* h1       = (_Float16*)alloc((size_t)20447232 * 2);     // [64][96][208][16]
  _Float16* h2       = (_Float16*)alloc((size_t)20447232 * 2);
  float*    cost_out = (float*)alloc((size_t)1277952 * 4);         // [64][96][208]
  float*    disp_low = (float*)alloc((size_t)19968 * 4);
  float*    disp0    = (float*)alloc((size_t)179712 * 4);
  float*    x2       = (float*)alloc((size_t)2 * 179712 * 4);
  float*    e1b      = (float*)alloc((size_t)718848 * 4);          // 16x144x312
  float*    e2b      = (float*)alloc((size_t)359424 * 4);          // 32x72x156
  float*    e3b      = (float*)alloc((size_t)179712 * 4);          // 64x36x78
  float*    u1b      = (float*)alloc((size_t)359424 * 4);          // 32x72x156
  float*    u2b      = (float*)alloc((size_t)718848 * 4);          // 16x144x312
  float*    u3b      = (float*)alloc((size_t)2875392 * 4);         // 16x288x624
  float*    refb     = (float*)alloc((size_t)179712 * 4);

  const int T = 256;
  auto nb = [](int n) { return (n + 255) / 256; };

  // Weight repack into WMMA A-fragment layouts
  repack_w1<<<nb(27648), T, 0, stream>>>(m_w1, wA1);
  repack_w2<<<nb(7168), T, 0, stream>>>(m_w2, wA2);

  // Feature extraction (left, then right) -> f16 [h][w][c]
  feat_conv1<<<nb(638976), T, 0, stream>>>(l0_img, f_w1, f_b1, fA);
  conv2d_generic<<<nb(638976), T, 0, stream>>>(fA, f_w2, f_b2, fB,
                                               32, 32, 96, 208, 96, 208, 1, 1, 0, 1);
  feat_conv3_f16<<<nb(638976), T, 0, stream>>>(fB, f_w3, f_b3, l0f);

  feat_conv1<<<nb(638976), T, 0, stream>>>(r0_img, f_w1, f_b1, fA);
  conv2d_generic<<<nb(638976), T, 0, stream>>>(fA, f_w2, f_b2, fB,
                                               32, 32, 96, 208, 96, 208, 1, 1, 0, 1);
  feat_conv3_f16<<<nb(638976), T, 0, stream>>>(fB, f_w3, f_b3, r0f);

  // 3D cost aggregation (WMMA): 19968 wave macro-tiles (16x64 each),
  // 8 waves/block -> 2496 blocks
  cost_conv1_wmma<<<2496, 256, 0, stream>>>(l0f, r0f, wA1, m_b1, h1);
  cost_conv2_wmma<<<2496, 256, 0, stream>>>(h1, wA2, m_b2, h2);
  cost_conv3<<<nb(1277952), T, 0, stream>>>(h2, m_w3, m_b3, cost_out);

  // Disparity regression + x3 upsample
  softmax_expect<<<nb(19968), T, 0, stream>>>(cost_out, disp_low);
  upsample3_linear<<<nb(179712), T, 0, stream>>>(disp_low, disp0);
  pack2<<<nb(179712), T, 0, stream>>>(disp0, d1_gt, x2);

  // Refinement U-Net (direct f32; ~2 GFLOP total)
  conv2d_generic<<<nb(718848), T, 0, stream>>>(x2, r_ew1, r_eb1, e1b,
                                               2, 16, 288, 624, 144, 312, 2, 0, 0, 1);
  conv2d_generic<<<nb(359424), T, 0, stream>>>(e1b, r_ew2, r_eb2, e2b,
                                               16, 32, 144, 312, 72, 156, 2, 0, 0, 1);
  conv2d_generic<<<nb(179712), T, 0, stream>>>(e2b, r_ew3, r_eb3, e3b,
                                               32, 64, 72, 156, 36, 78, 2, 0, 0, 1);
  conv2d_generic<<<nb(359424), T, 0, stream>>>(e3b, r_dw1, r_db1, u1b,
                                               64, 32, 36, 78, 72, 156, 1, 1, 1, 1);
  conv2d_generic<<<nb(718848), T, 0, stream>>>(u1b, r_dw2, r_db2, u2b,
                                               32, 16, 72, 156, 144, 312, 1, 1, 1, 1);
  conv2d_generic<<<nb(2875392), T, 0, stream>>>(u2b, r_dw3, r_db3, u3b,
                                                16, 16, 144, 312, 288, 624, 1, 1, 1, 1);
  conv2d_generic<<<nb(179712), T, 0, stream>>>(u3b, r_ow, r_ob, refb,
                                               16, 1, 288, 624, 288, 624, 1, 1, 0, 0);
  add_out<<<nb(179712), T, 0, stream>>>(refb, disp0, out);
}
